// GraphAttention_83494164234359
// MI455X (gfx1250) — matrix operand order repaired and checked by reading, not compiled, not run on hardware
//
#include <hip/hip_runtime.h>

typedef __attribute__((ext_vector_type(16))) __bf16 v16bf;
typedef __attribute__((ext_vector_type(8)))  __bf16 v8bf;
typedef __attribute__((ext_vector_type(8)))  float  v8f;
typedef __attribute__((ext_vector_type(4)))  unsigned int u32x4;
typedef __attribute__((ext_vector_type(8)))  int          i32x8;
typedef __attribute__((ext_vector_type(4)))  int          i32x4;

#define B_ 4
#define N_ 2048
#define F_ 256
#define H_ 8
#define U_ 128
#define NEGC (-1.0e10f)
#define LEAK 0.2f
#define VSTRIDE 40   // 32 + 8 bf16 pad -> 80B row stride, conflict-free ds_load_b128

// ---------------------------------------------------------------------------
// TDM: stage one 128-row x 64-byte tile (32 j-columns of featsT, bf16) into
// LDS, with 16B padding after every 64B row (matches VSTRIDE layout).
// 2D descriptor, data_size = 4B dwords.  Issued by one wave per workgroup.
// ---------------------------------------------------------------------------
__device__ __forceinline__ void tdm_stage_vtile(unsigned lds_byte_addr,
                                                const __bf16* gsrc) {
  unsigned long long ga = (unsigned long long)(uintptr_t)gsrc;
  u32x4 g0;
  g0[0] = 1u;                                        // count=1, user D#
  g0[1] = lds_byte_addr;                             // lds_addr [63:32]
  g0[2] = (unsigned)(ga & 0xffffffffu);              // global_addr lo
  g0[3] = (unsigned)((ga >> 32) & 0x01ffffffu)       // global_addr hi (57b)
          | (2u << 30);                              // type=2 ("image")
  const unsigned td0 = 1024;                         // tensor_dim0 (dwords/row)
  const unsigned td1 = 128;                          // tensor_dim1 (rows)
  const unsigned tile0 = 16;                         // tile_dim0 (16 dw = 64B)
  const unsigned tile1 = 128;                        // tile_dim1 (rows)
  const unsigned str0 = 1024;                        // row stride (dwords)
  i32x8 g1;
  g1[0] = (int)((2u << 16)                           // data_size = 4B
                | (1u << 20)                         // pad_enable
                | (3u << 22)                         // pad_interval: 16 dwords
                | (3u << 25));                       // pad_amount: 4 dwords
  g1[1] = (int)((td0 & 0xffffu) << 16);              // tensor_dim0[15:0]
  g1[2] = (int)(((td0 >> 16) & 0xffffu) | ((td1 & 0xffffu) << 16));
  g1[3] = (int)(((td1 >> 16) & 0xffffu) | (tile0 << 16));
  g1[4] = (int)(tile1);                              // tile_dim1 ; tile_dim2=0
  g1[5] = (int)str0;                                 // dim0_stride[31:0]
  g1[6] = 0;                                         // stride hi / dim1_stride
  g1[7] = 0;
  i32x4 z4 = {0, 0, 0, 0};                           // groups 2/3 unused (2D)
  i32x8 z8 = {0, 0, 0, 0, 0, 0, 0, 0};
  __builtin_amdgcn_tensor_load_to_lds(g0, g1, z4, z4, z8, 0);
}

// ---------------------------------------------------------------------------
// Kernel 0: W [H,F,U] f32  ->  Wt [H,U,F] bf16
// ---------------------------------------------------------------------------
__global__ __launch_bounds__(256) void gat_wt(const float* __restrict__ W,
                                              __bf16* __restrict__ Wt) {
  int idx = blockIdx.x * 256 + threadIdx.x;       // H*U*F = 262144 threads
  if (idx >= H_ * U_ * F_) return;
  int f  = idx % F_;
  int hu = idx / F_;
  int u  = hu % U_;
  int h  = hu / U_;
  Wt[idx] = (__bf16)W[((size_t)h * F_ + f) * U_ + u];
}

// ---------------------------------------------------------------------------
// Kernel 1: feats = x @ W per (b,h). Each wave: 16 rows x 128 cols, K=256.
// Output stored TRANSPOSED: featsT [B,H,U,N] bf16.
// ---------------------------------------------------------------------------
__global__ __launch_bounds__(256) void gat_proj(const float*  __restrict__ x,
                                                const __bf16* __restrict__ Wt,
                                                __bf16* __restrict__ featsT) {
  const int lane = threadIdx.x & 31;
  const int wave = threadIdx.x >> 5;
  const int nc = blockIdx.x & 15;                 // 16 chunks of 128 rows
  const int bh = blockIdx.x >> 4;                 // b*H + h
  const int b  = bh >> 3, h = bh & 7;
  const int n0 = nc * 128 + wave * 16;
  const int r  = lane & 15;
  const int kh = lane >> 4;

  const float*  xrow  = x + ((size_t)b * N_ + n0 + r) * F_;
  const __bf16* wbase = Wt + (size_t)h * U_ * F_;

  v8f acc[8] = {};

  for (int k0 = 0; k0 < F_; k0 += 32) {
    union { float4 q[4]; float f[16]; } X;
    X.q[0] = *(const float4*)(xrow + k0 + 8 * kh);
    X.q[1] = *(const float4*)(xrow + k0 + 8 * kh + 4);
    X.q[2] = *(const float4*)(xrow + k0 + 16 + 8 * kh);
    X.q[3] = *(const float4*)(xrow + k0 + 16 + 8 * kh + 4);
    v16bf av;
#pragma unroll
    for (int i = 0; i < 16; ++i) av[i] = (__bf16)X.f[i];

    v16bf bv[8];
#pragma unroll
    for (int ut = 0; ut < 8; ++ut) {
      const __bf16* wp = wbase + (size_t)(ut * 16 + r) * F_ + k0 + 8 * kh;
      v8bf lo = *(const v8bf*)wp;
      v8bf hi = *(const v8bf*)(wp + 16);
#pragma unroll
      for (int i = 0; i < 8; ++i) { bv[ut][i] = lo[i]; bv[ut][i + 8] = hi[i]; }
    }
#pragma unroll
    for (int ut = 0; ut < 8; ++ut)
      acc[ut] = __builtin_amdgcn_wmma_f32_16x16x32_bf16(
          false, av, false, bv[ut], (short)0, acc[ut], false, false);
  }

#pragma unroll
  for (int ut = 0; ut < 8; ++ut) {
    int u = ut * 16 + r;
    v8bf st;
#pragma unroll
    for (int v = 0; v < 8; ++v) st[v] = (__bf16)acc[ut][v];
    *(v8bf*)(featsT + ((size_t)bh * U_ + u) * N_ + n0 + 8 * kh) = st;
  }
}

// ---------------------------------------------------------------------------
// Kernel 2: s[bh,n] = feats . a_self ; t[bh,n] = feats . a_neigh
// ---------------------------------------------------------------------------
__global__ __launch_bounds__(256) void gat_st(const __bf16* __restrict__ featsT,
                                              const float* __restrict__ a_self,
                                              const float* __restrict__ a_neigh,
                                              float* __restrict__ svec,
                                              float* __restrict__ tvec) {
  int idx = blockIdx.x * 256 + threadIdx.x;       // B*H*N = 65536
  if (idx >= B_ * H_ * N_) return;
  int n  = idx % N_;
  int bh = idx / N_;
  int h  = bh & 7;
  const __bf16* fp = featsT + (size_t)bh * U_ * N_ + n;
  const float* as = a_self + h * U_;
  const float* an = a_neigh + h * U_;
  float s = 0.0f, t = 0.0f;
#pragma unroll 4
  for (int u = 0; u < U_; ++u) {
    float fv = (float)fp[(size_t)u * N_];
    s += fv * as[u];
    t += fv * an[u];
  }
  svec[idx] = s;
  tvec[idx] = t;
}

// ---------------------------------------------------------------------------
// Kernel 3: fused flash-GAT. Workgroup = 8 waves, each owns 16 query rows of
// one (b,h). Each 32x128 V chunk is DMA'd into LDS by the Tensor Data Mover
// (one descriptor per chunk, double-buffered, issued by wave 0, synced with
// s_wait_tensorcnt + barrier); TDM padding gives the conflict-free 80B rows.
// Streaming softmax over j in chunks of 32 (= WMMA K), P x V via
// v_wmma_f32_16x16x32_bf16, never materializing the NxN score matrix.
// ---------------------------------------------------------------------------
__global__ __launch_bounds__(256) void gat_attn(const __bf16* __restrict__ featsT,
                                                const float* __restrict__ svec,
                                                const float* __restrict__ tvec,
                                                const float* __restrict__ adj,
                                                const float* __restrict__ bias,
                                                float* __restrict__ out) {
  __shared__ __bf16 vtile[2][U_ * VSTRIDE];       // 2 x 10 KB

  const int tid  = threadIdx.x;
  const int lane = tid & 31;
  const int wave = tid >> 5;
  const int nc = blockIdx.x & 15;
  const int bh = blockIdx.x >> 4;
  const int b  = bh >> 3, h = bh & 7;
  const int i0 = nc * 128 + wave * 16;
  const int r  = lane & 15;
  const int kh = lane >> 4;

  const size_t bhN   = (size_t)bh * N_;
  const float  s_row = svec[bhN + i0 + r];
  const float* trow   = tvec + bhN;
  const float* adjrow = adj + ((size_t)b * N_ + (i0 + r)) * N_;
  const __bf16* vbase = featsT + (size_t)bh * U_ * N_;

  // LDS byte addresses of the two buffers (low 32 bits of generic address)
  const unsigned lbuf0 = (unsigned)(uintptr_t)&vtile[0][0];
  const unsigned lbuf1 = (unsigned)(uintptr_t)&vtile[1][0];

  v8f acc[8] = {};
  float m_run = -3.0e38f;
  float l_run = 0.0f;

  // prologue: DMA chunk 0 into buffer 0
  if (wave == 0) tdm_stage_vtile(lbuf0, vbase);

  int p = 0;
  for (int jc = 0; jc < N_; jc += 32) {
    // ---- scores for this lane's row (A-fragment register layout) ----
    // (pure VALU + global loads: overlaps the in-flight TDM for chunk jc)
    const int o0 = jc + 8 * kh;
    const int o1 = jc + 16 + 8 * kh;
    union { float4 q[4]; float f[16]; } T, A;
    T.q[0] = *(const float4*)(trow + o0);
    T.q[1] = *(const float4*)(trow + o0 + 4);
    T.q[2] = *(const float4*)(trow + o1);
    T.q[3] = *(const float4*)(trow + o1 + 4);
    A.q[0] = *(const float4*)(adjrow + o0);
    A.q[1] = *(const float4*)(adjrow + o0 + 4);
    A.q[2] = *(const float4*)(adjrow + o1);
    A.q[3] = *(const float4*)(adjrow + o1 + 4);

    float e[16];
    float mloc = -3.0e38f;
#pragma unroll
    for (int i = 0; i < 16; ++i) {
      float ev = s_row + T.f[i];
      ev = ev > 0.0f ? ev : LEAK * ev;          // LeakyReLU(0.2)
      ev += NEGC * (1.0f - A.f[i]);             // additive adjacency mask
      e[i] = ev;
      mloc = fmaxf(mloc, ev);
    }
    mloc = fmaxf(mloc, __shfl_xor(mloc, 16, 32));
    float newm  = fmaxf(m_run, mloc);
    float scale = __expf(m_run - newm);

    float psum = 0.0f;
    v16bf pa;
#pragma unroll
    for (int i = 0; i < 16; ++i) {
      float pv = __expf(e[i] - newm);
      psum += pv;
      pa[i] = (__bf16)pv;
    }
    psum += __shfl_xor(psum, 16, 32);
    l_run = l_run * scale + psum;
    m_run = newm;

    // rescale accumulator: C-layout row for vgpr v is (8*kh + v)
#pragma unroll
    for (int v = 0; v < 8; ++v) {
      float sv = __shfl(scale, 8 * kh + v, 32);
#pragma unroll
      for (int ut = 0; ut < 8; ++ut) acc[ut][v] *= sv;
    }

    // ---- publish chunk jc, then kick off DMA for chunk jc+32 ----
    if (wave == 0) __builtin_amdgcn_s_wait_tensorcnt(0);
    __syncthreads();     // staging visible; all waves done reading buf p^1
    if (wave == 0 && jc + 32 < N_)
      tdm_stage_vtile(p ? lbuf0 : lbuf1, vbase + jc + 32);

    // ---- load all 8 V fragments from LDS, then 8 WMMAs back-to-back ----
    const __bf16* lb = &vtile[p][0] + 8 * kh;
    v16bf bv[8];
#pragma unroll
    for (int ut = 0; ut < 8; ++ut) {
      const __bf16* vp = lb + (ut * 16 + r) * VSTRIDE;
      v8bf lo = *(const v8bf*)vp;
      v8bf hi = *(const v8bf*)(vp + 16);
#pragma unroll
      for (int i = 0; i < 8; ++i) { bv[ut][i] = lo[i]; bv[ut][i + 8] = hi[i]; }
    }
#pragma unroll
    for (int ut = 0; ut < 8; ++ut)
      acc[ut] = __builtin_amdgcn_wmma_f32_16x16x32_bf16(
          false, pa, false, bv[ut], (short)0, acc[ut], false, false);

    p ^= 1;
  }

  // epilogue: normalize, bias, ReLU, concat-heads layout [B,N,H*U]
#pragma unroll
  for (int v = 0; v < 8; ++v) {
    float lv  = __shfl(l_run, 8 * kh + v, 32);
    float inv = 1.0f / lv;
    int   row = i0 + 8 * kh + v;
#pragma unroll
    for (int ut = 0; ut < 8; ++ut) {
      int   u = ut * 16 + r;
      float o = acc[ut][v] * inv + bias[h * U_ + u];
      o = o > 0.0f ? o : 0.0f;
      out[((size_t)b * N_ + row) * (H_ * U_) + h * U_ + u] = o;
    }
  }
}

// ---------------------------------------------------------------------------
extern "C" void kernel_launch(void* const* d_in, const int* in_sizes, int n_in,
                              void* d_out, int out_size, void* d_ws, size_t ws_size,
                              hipStream_t stream) {
  const float* x       = (const float*)d_in[0];   // [B,N,F]
  const float* adj     = (const float*)d_in[1];   // [B,N,N]
  const float* W       = (const float*)d_in[2];   // [H,F,U]
  const float* a_self  = (const float*)d_in[3];   // [H,U]
  const float* a_neigh = (const float*)d_in[4];   // [H,U]
  const float* bias    = (const float*)d_in[5];   // [H,U]
  float* out = (float*)d_out;                     // [B,N,H*U] f32

  char* ws = (char*)d_ws;
  const size_t WT_BYTES = (size_t)H_ * U_ * F_ * sizeof(__bf16);      // 512 KB
  const size_t FT_BYTES = (size_t)B_ * H_ * U_ * N_ * sizeof(__bf16); // 16 MB
  const size_t S_BYTES  = (size_t)B_ * H_ * N_ * sizeof(float);       // 256 KB
  __bf16* Wt     = (__bf16*)ws;
  __bf16* featsT = (__bf16*)(ws + WT_BYTES);
  float*  svec   = (float*)(ws + WT_BYTES + FT_BYTES);
  float*  tvec   = (float*)(ws + WT_BYTES + FT_BYTES + S_BYTES);

  gat_wt  <<<(H_ * U_ * F_ + 255) / 256, 256, 0, stream>>>(W, Wt);
  gat_proj<<<B_ * H_ * (N_ / 128),       256, 0, stream>>>(x, Wt, featsT);
  gat_st  <<<(B_ * H_ * N_ + 255) / 256, 256, 0, stream>>>(featsT, a_self,
                                                           a_neigh, svec, tvec);
  gat_attn<<<B_ * H_ * (N_ / 128),       256, 0, stream>>>(featsT, svec, tvec,
                                                           adj, bias, out);
}
